// LSTMDecoder_29978871726104
// MI455X (gfx1250) — matrix operand order repaired and checked by reading, not compiled, not run on hardware
//
#include <hip/hip_runtime.h>
#include <hip/hip_bf16.h>

typedef __attribute__((ext_vector_type(16))) _Float16 v16h;
typedef __attribute__((ext_vector_type(8)))  float    v8f;

#define HID    128
#define LATENT 32
#define NVAR   9
#define NGATE  4
#define NWAVE  8
#define HSTR   136   // halfs per row of h LDS buffers: 272B stride, 16B aligned, bank-conflict free
#define PSTR   40    // halfs per row of prev-output staging buffer

__device__ __forceinline__ v8f wmma_f16(v16h a, v16h b, v8f c) {
  return __builtin_amdgcn_wmma_f32_16x16x32_f16(false, a, false, b, (short)0, c, false, false);
}

__device__ __forceinline__ float fsigmoid(float x) {
  return 1.0f / (1.0f + __expf(-x));
}
__device__ __forceinline__ float ftanh_(float x) {
  x = fminf(fmaxf(x, -15.0f), 15.0f);
  float e = __expf(2.0f * x);
  return (e - 1.0f) / (e + 1.0f);
}

// ---- B-fragment builders: 32(K)x16(N) f16 fragment of row-major f32 W ----
// Layout: lanes 0-15 hold K=0..15 of column N=lane; lanes 16-31 hold K=16..31 of N=lane-16.

// General (bounds-checked, arbitrary k offset) builder.
__device__ __forceinline__ v16h build_bfrag(const float* W, int ld, int nbase,
                                            int kb, int koff, int kend, int nend) {
  const int lane = threadIdx.x & 31;
  const int n    = nbase + (lane & 15);
  const int k0   = kb + ((lane >> 4) << 4);
  v16h r;
#pragma unroll
  for (int j = 0; j < 16; ++j) {
    int k = k0 + j + koff;
    float v = (((unsigned)k < (unsigned)kend) && (n < nend)) ? W[n * ld + k] : 0.0f;
    r[j] = (_Float16)v;
  }
  return r;
}

// Fast builder: full K range valid, rows 16B aligned (ld % 4 == 0) -> float4 loads.
__device__ __forceinline__ v16h build_bfrag_vec(const float* W, int ld, int nbase, int kb) {
  const int lane = threadIdx.x & 31;
  const int n    = nbase + (lane & 15);
  const int k0   = kb + ((lane >> 4) << 4);
  const float4* p = (const float4*)(W + n * ld + k0);
  v16h r;
#pragma unroll
  for (int q = 0; q < 4; ++q) {
    float4 f = p[q];
    r[4 * q + 0] = (_Float16)f.x;
    r[4 * q + 1] = (_Float16)f.y;
    r[4 * q + 2] = (_Float16)f.z;
    r[4 * q + 3] = (_Float16)f.w;
  }
  return r;
}

// ---- A-fragment loader: 16(M)x32(K) f16 from row-major LDS buffer ----
// lane<16: M=lane, halfs 0-15 = K kb+0..15; lane>=16: M=lane-16, K kb+16..31.
__device__ __forceinline__ v16h load_afrag(const _Float16* buf, int stride, int kb) {
  const int lane = threadIdx.x & 31;
  const int m    = lane & 15;
  const int hi   = lane >> 4;
  const _Float16* p = buf + m * stride + kb + hi * 16;
  v16h r;
#pragma unroll
  for (int j = 0; j < 8; ++j) r[j] = p[j];
#pragma unroll
  for (int j = 0; j < 8; ++j) r[8 + j] = p[8 + j];
  return r;
}

__global__ __launch_bounds__(256, 1)
void lstm_decoder_kernel(
    const float* __restrict__ z,      const int*   __restrict__ scenario,
    const int*   __restrict__ seqlen, const float* __restrict__ emb,
    const float* __restrict__ W_init, const float* __restrict__ b_init,
    const float* __restrict__ W_ih0,  const float* __restrict__ W_hh0,
    const float* __restrict__ b_ih0,  const float* __restrict__ b_hh0,
    const float* __restrict__ W_ih1,  const float* __restrict__ W_hh1,
    const float* __restrict__ b_ih1,  const float* __restrict__ b_hh1,
    const float* __restrict__ W_out,  const float* __restrict__ b_out,
    float* __restrict__ out)
{
  __shared__ _Float16 hbuf0[16 * HSTR];               //  4.25 KB
  __shared__ _Float16 hbuf1[16 * HSTR];               //  4.25 KB
  __shared__ _Float16 prevbuf[16 * PSTR];             //  1.25 KB
  __shared__ v16h     Bhh1L[NWAVE][NGATE][4][32];     // 128 KB, fragment-ready layout
  __shared__ v16h     BoutL[4][32];                   //   4 KB
  __shared__ v8f      seCL[NWAVE][NGATE][32];         //  32 KB

  const int b0   = blockIdx.x * 16;
  const int wave = threadIdx.x >> 5;
  const int lane = threadIdx.x & 31;
  const int mrow = lane & 15;           // A-frag row / C-frag column index
  const int hi   = lane >> 4;
  const int col  = 16 * wave + mrow;    // this wave+lane's h column
  const int T    = seqlen[0];

  // zero prev staging buffer; columns >= NVAR stay zero forever
  for (int i = threadIdx.x; i < 16 * PSTR; i += 256) prevbuf[i] = (_Float16)0.0f;

  // ---- build [z | se] init A fragments (K = 160 = 5 chunks of 32) ----
  const int scen = scenario[b0 + mrow];
  v16h Ainit[5];
#pragma unroll
  for (int c = 0; c < 5; ++c) {
    v16h a;
#pragma unroll
    for (int j = 0; j < 16; ++j) {
      int k = 32 * c + hi * 16 + j;
      float v = (k < LATENT) ? z[(b0 + mrow) * LATENT + k]
                             : emb[scen * HID + (k - LATENT)];
      a[j] = (_Float16)v;
    }
    Ainit[c] = a;
  }

  // ---- h/c init: [z|se] @ W_init^T + b_init ; parts p = h0,c0,h1,c1 ----
  v8f c0reg, c1reg;
#pragma unroll
  for (int p = 0; p < 4; ++p) {
    v8f acc = {};
#pragma unroll
    for (int c = 0; c < 5; ++c) {
      v16h b = build_bfrag_vec(W_init, LATENT + HID, 128 * p + 16 * wave, 32 * c);
      acc = wmma_f16(Ainit[c], b, acc);
    }
    float bias = b_init[128 * p + col];
    if (p == 0) {
#pragma unroll
      for (int r = 0; r < 8; ++r) hbuf0[(r + 8 * hi) * HSTR + col] = (_Float16)(acc[r] + bias);
    } else if (p == 1) {
#pragma unroll
      for (int r = 0; r < 8; ++r) c0reg[r] = acc[r] + bias;
    } else if (p == 2) {
#pragma unroll
      for (int r = 0; r < 8; ++r) hbuf1[(r + 8 * hi) * HSTR + col] = (_Float16)(acc[r] + bias);
    } else {
#pragma unroll
      for (int r = 0; r < 8; ++r) c1reg[r] = acc[r] + bias;
    }
  }

  // ---- time-invariant scenario contribution to layer-0 gates (-> LDS) + biases ----
  float bias1[NGATE];
#pragma unroll
  for (int g = 0; g < NGATE; ++g) {
    const int ng = 128 * g + 16 * wave;
    v8f acc = {};
#pragma unroll
    for (int c = 1; c < 5; ++c) {  // se sits at K=32..159 ; W_ih0 column = K - 32 + 9
      v16h b = build_bfrag(W_ih0, NVAR + HID, ng, 32 * c, NVAR - LATENT, NVAR + HID, 4 * HID);
      acc = wmma_f16(Ainit[c], b, acc);
    }
    float bias = b_ih0[ng + mrow] + b_hh0[ng + mrow];
#pragma unroll
    for (int r = 0; r < 8; ++r) acc[r] += bias;
    seCL[wave][g][lane] = acc;
    bias1[g] = b_ih1[ng + mrow] + b_hh1[ng + mrow];
  }

  // ---- LDS-resident fragments (built before the big register arrays go live) ----
#pragma unroll
  for (int g = 0; g < NGATE; ++g) {
    const int ng = 128 * g + 16 * wave;
#pragma unroll
    for (int c = 0; c < 4; ++c)
      Bhh1L[wave][g][c][lane] = build_bfrag_vec(W_hh1, HID, ng, 32 * c);
  }
  float boutreg = 0.0f;
  if (wave == 0) {
#pragma unroll
    for (int c = 0; c < 4; ++c)
      BoutL[c][lane] = build_bfrag(W_out, HID, 0, 32 * c, 0, HID, NVAR);  // N 9..15 = 0
    if (mrow < NVAR) boutreg = b_out[mrow];
  }

  // ---- scalar-path prev-input slice (K 9..31 = 0) ----
  v16h Bp[NGATE];
#pragma unroll
  for (int g = 0; g < NGATE; ++g)
    Bp[g] = build_bfrag(W_ih0, NVAR + HID, 128 * g + 16 * wave, 0, 0, NVAR, 4 * HID);

  // ---- register-resident weight fragments, filled last (lowest transient pressure) ----
  v16h Bhh0[NGATE][4], Bih1[NGATE][4];
#pragma unroll
  for (int g = 0; g < NGATE; ++g) {
    const int ng = 128 * g + 16 * wave;
#pragma unroll
    for (int c = 0; c < 4; ++c) {
      Bhh0[g][c] = build_bfrag_vec(W_hh0, HID, ng, 32 * c);
      Bih1[g][c] = build_bfrag_vec(W_ih1, HID, ng, 32 * c);
    }
  }

  __syncthreads();   // init h/prev buffers + LDS fragments visible

  // ---- prologue: hoisted recurrent contributions for step 0 ----
  v8f g0acc[NGATE], g1acc[NGATE];
  {
    v16h A0i[4], A1i[4];
#pragma unroll
    for (int c = 0; c < 4; ++c) {
      A0i[c] = load_afrag(hbuf0, HSTR, 32 * c);
      A1i[c] = load_afrag(hbuf1, HSTR, 32 * c);
    }
#pragma unroll
    for (int g = 0; g < NGATE; ++g) {
      v8f a0 = seCL[wave][g][lane];
      v8f a1;
#pragma unroll
      for (int r = 0; r < 8; ++r) a1[r] = bias1[g];
#pragma unroll
      for (int c = 0; c < 4; ++c) {
        a0 = wmma_f16(A0i[c], Bhh0[g][c], a0);
        a1 = wmma_f16(A1i[c], Bhh1L[wave][g][c][lane], a1);
      }
      g0acc[g] = a0;
      g1acc[g] = a1;
    }
  }

  for (int t = 0; t < T; ++t) {
    // ---- layer 0: only the K=9 prev-feedback GEMM remains on the chain ----
    v16h Ap = load_afrag(prevbuf, PSTR, 0);
    v8f g0[NGATE];
#pragma unroll
    for (int g = 0; g < NGATE; ++g) g0[g] = wmma_f16(Ap, Bp[g], g0acc[g]);

    float h0n[8];
#pragma unroll
    for (int r = 0; r < 8; ++r) {
      float ci = fsigmoid(g0[0][r]);
      float cf = fsigmoid(g0[1][r]);
      float cg = ftanh_(g0[2][r]);
      float co = fsigmoid(g0[3][r]);
      float cn = cf * c0reg[r] + ci * cg;
      c0reg[r] = cn;
      h0n[r] = co * ftanh_(cn);
    }
#pragma unroll
    for (int r = 0; r < 8; ++r) hbuf0[(r + 8 * hi) * HSTR + col] = (_Float16)h0n[r];
    __syncthreads();   // B1: h0_new visible

    // ---- layer 1 input GEMM (register B) + hoisted layer-0 recurrent ----
    v16h A0n[4];
#pragma unroll
    for (int c = 0; c < 4; ++c) A0n[c] = load_afrag(hbuf0, HSTR, 32 * c);

    v8f g1[NGATE];
#pragma unroll
    for (int g = 0; g < NGATE; ++g) {
      v8f acc = g1acc[g];
#pragma unroll
      for (int c = 0; c < 4; ++c) acc = wmma_f16(A0n[c], Bih1[g][c], acc);
      g1[g] = acc;
    }
#pragma unroll
    for (int g = 0; g < NGATE; ++g) {          // next-step layer-0 recurrent (reuses A0n)
      v8f acc = seCL[wave][g][lane];
#pragma unroll
      for (int c = 0; c < 4; ++c) acc = wmma_f16(A0n[c], Bhh0[g][c], acc);
      g0acc[g] = acc;
    }

    float h1n[8];
#pragma unroll
    for (int r = 0; r < 8; ++r) {
      float ci = fsigmoid(g1[0][r]);
      float cf = fsigmoid(g1[1][r]);
      float cg = ftanh_(g1[2][r]);
      float co = fsigmoid(g1[3][r]);
      float cn = cf * c1reg[r] + ci * cg;
      c1reg[r] = cn;
      h1n[r] = co * ftanh_(cn);
    }
#pragma unroll
    for (int r = 0; r < 8; ++r) hbuf1[(r + 8 * hi) * HSTR + col] = (_Float16)h1n[r];
    __syncthreads();   // B2: h1_new visible

    // ---- hoisted layer-1 recurrent for next step (B streamed from LDS) +
    //      output projection / prev staging (wave 0, overlapped) ----
    v16h A1n[4];
#pragma unroll
    for (int c = 0; c < 4; ++c) A1n[c] = load_afrag(hbuf1, HSTR, 32 * c);

#pragma unroll
    for (int g = 0; g < NGATE; ++g) {
      v8f acc;
#pragma unroll
      for (int r = 0; r < 8; ++r) acc[r] = bias1[g];
#pragma unroll
      for (int c = 0; c < 4; ++c) acc = wmma_f16(A1n[c], Bhh1L[wave][g][c][lane], acc);
      g1acc[g] = acc;
    }

    if (wave == 0) {
      v8f oacc = {};
#pragma unroll
      for (int c = 0; c < 4; ++c) oacc = wmma_f16(A1n[c], BoutL[c][lane], oacc);
#pragma unroll
      for (int r = 0; r < 8; ++r) {
        int m = r + 8 * hi;
        float v = (mrow < NVAR) ? (oacc[r] + boutreg) : 0.0f;
        prevbuf[m * PSTR + mrow] = (_Float16)v;
        if (mrow < NVAR)
          out[(size_t)(b0 + m) * (size_t)T * NVAR + (size_t)t * NVAR + mrow] = v;
      }
    }
    __syncthreads();   // B3: prevbuf ready for next iteration
  }
}

extern "C" void kernel_launch(void* const* d_in, const int* in_sizes, int n_in,
                              void* d_out, int out_size, void* d_ws, size_t ws_size,
                              hipStream_t stream) {
  const int B = in_sizes[0] / LATENT;
  dim3 grid(B / 16), block(256);
  lstm_decoder_kernel<<<grid, block, 0, stream>>>(
      (const float*)d_in[0],  (const int*)d_in[1],   (const int*)d_in[2],
      (const float*)d_in[3],  (const float*)d_in[4], (const float*)d_in[5],
      (const float*)d_in[6],  (const float*)d_in[7], (const float*)d_in[8],
      (const float*)d_in[9],  (const float*)d_in[10],(const float*)d_in[11],
      (const float*)d_in[12], (const float*)d_in[13],(const float*)d_in[14],
      (const float*)d_in[15], (float*)d_out);
}
